// Solograph_HeteroGNN_5463198401035
// MI455X (gfx1250) — matrix operand order repaired and checked by reading, not compiled, not run on hardware
//
#include <hip/hip_runtime.h>
#include <hip/hip_bf16.h>
#include <math.h>
#include <stdint.h>

#define N_NODES 10000
#define PAD_ROWS 10112          // ceil(10000/128)*128, async-copy overread slack
#define HIDV    128
#define NEDGE   64000
#define SLOPEV  0.2f
#define EPSV    1e-16f

typedef __attribute__((ext_vector_type(16))) __bf16 v16bf;
typedef __attribute__((ext_vector_type(8)))  float  v8f;

// ---------- helpers ----------
__device__ __forceinline__ unsigned short f2bf(float f) {
    unsigned int u = __float_as_uint(f);
    u = u + 0x7FFFu + ((u >> 16) & 1u);   // round-to-nearest-even
    return (unsigned short)(u >> 16);
}
// order-preserving float<->uint encoding for atomicMax on floats
__device__ __forceinline__ unsigned int encOrd(float f) {
    unsigned int u = __float_as_uint(f);
    return (u & 0x80000000u) ? ~u : (u | 0x80000000u);
}
__device__ __forceinline__ float decOrd(unsigned int e) {
    unsigned int u = (e & 0x80000000u) ? (e & 0x7FFFFFFFu) : ~e;
    return __uint_as_float(u);
}

// ---------- pre-pass: fp32 -> bf16 feature copy ----------
__global__ void cvt_x_kernel(const float* __restrict__ X, unsigned short* __restrict__ Xb, int n) {
    int i = blockIdx.x * blockDim.x + threadIdx.x;
    if (i < n) Xb[i] = f2bf(X[i]);
}

// ---------- pre-pass: fp32 W[k][n] -> bf16 Wt[n][k] ----------
__global__ void cvt_wt_kernel(const float* __restrict__ W, unsigned short* __restrict__ Wt) {
    int i = blockIdx.x * blockDim.x + threadIdx.x;
    if (i < 128 * 128) {
        int k = i >> 7, n = i & 127;
        Wt[n * 128 + k] = f2bf(W[i]);
    }
}

// ---------- GEMM: Y[rows x 128] = Xb[rows x 128](bf16) @ Wt^T(bf16), WMMA ----------
// A tile staged HBM->LDS via async-to-LDS; B fragments read straight from L2-resident Wt.
__global__ __launch_bounds__(256) void gemm_xw_kernel(const unsigned short* __restrict__ Xb,
                                                      const unsigned short* __restrict__ Wt,
                                                      float* __restrict__ Y,
                                                      int rows) {
    __shared__ unsigned short As[128 * 128]; // 32 KB: X tile, bf16, row-major [m][k]

    const int t = threadIdx.x;
    const int rowBase = blockIdx.x * 128;

    // async-stage 32 KB tile: 256 threads x 8 issues x 16 B
    {
        const char* gsrc = (const char*)(Xb + (size_t)rowBase * 128);
        unsigned ldsBase = (unsigned)(uintptr_t)(void*)As;  // low 32 bits of flat LDS ptr = LDS offset
        #pragma unroll
        for (int i = 0; i < 8; ++i) {
            unsigned o  = (unsigned)((i * 256 + t) * 16);
            unsigned la = ldsBase + o;
            asm volatile("global_load_async_to_lds_b128 %0, %1, %2"
                         :: "v"(la), "v"(o), "s"(gsrc) : "memory");
        }
        asm volatile("s_wait_asynccnt 0x0" ::: "memory");
    }
    __syncthreads();

    const int lane = t & 31;
    const int w    = t >> 5;     // wave 0..7 -> rows [16w, 16w+16)
    const int M    = lane & 15;
    const int half = lane >> 4;

    const unsigned int* A32 = (const unsigned int*)As;

    v8f acc[8];
    #pragma unroll
    for (int i = 0; i < 8; ++i) acc[i] = (v8f){0.f, 0.f, 0.f, 0.f, 0.f, 0.f, 0.f, 0.f};

    union FragA { v16bf v; unsigned int u[8]; };
    union FragB { v16bf v; uint4 q[2]; };

    const int mrow = w * 16 + M;
    #pragma unroll
    for (int ks = 0; ks < 4; ++ks) {
        const int k0 = ks * 32;
        // A fragment: 16-bit A 16x32 layout (ISA 7.12.2), packed 32-bit LDS reads
        FragA a;
        #pragma unroll
        for (int v = 0; v < 8; ++v) {
            int kb = (v < 4) ? (2 * v) : (16 + 2 * (v - 4));
            a.u[v] = A32[(mrow * 128 + k0 + kb + 8 * half) >> 1];
        }
        #pragma unroll
        for (int nt = 0; nt < 8; ++nt) {
            // B fragment: 32x16, lanes 0-15 K=0..7,16..23, lanes 16-31 K=8..15,24..31
            FragB b;
            const uint4* p = (const uint4*)(Wt + (size_t)(nt * 16 + M) * 128 + k0 + 8 * half);
            b.q[0] = p[0];
            b.q[1] = p[2];
            acc[nt] = __builtin_amdgcn_wmma_f32_16x16x32_bf16(
                false, a.v, false, b.v, (short)0, acc[nt], false, false);
        }
    }
    // store: C/D layout - VGPR r: lanes0-15 M=r, lanes16-31 M=8+r; N=lane%16
    #pragma unroll
    for (int nt = 0; nt < 8; ++nt) {
        #pragma unroll
        for (int r = 0; r < 8; ++r) {
            int grow = rowBase + w * 16 + r + 8 * half;
            if (grow < rows) Y[grow * 128 + nt * 16 + M] = acc[nt][r];
        }
    }
}

// ---------- per-instance init: segment max (enc(-inf)) and denominators ----------
__global__ void init_md_kernel(unsigned int* __restrict__ mEnc, float* __restrict__ den, int n) {
    int i = blockIdx.x * blockDim.x + threadIdx.x;
    if (i < n) { mEnc[i] = 0x007FFFFFu; den[i] = 0.f; } // enc(-inf)
}

// ---------- E1: logits + segment max (wave per edge) ----------
__global__ void e1_kernel(const int* __restrict__ ei,
                          const float* __restrict__ xl, const float* __restrict__ xr,
                          const float* __restrict__ att,
                          float* __restrict__ logit, unsigned int* __restrict__ mEnc,
                          int E, int Nn) {
    int wid  = (blockIdx.x * blockDim.x + threadIdx.x) >> 5;
    int lane = threadIdx.x & 31;
    int T = E + Nn;
    if (wid >= T) return;
    int src, dst;
    if (wid < E) { src = ei[wid]; dst = ei[E + wid]; }
    else         { src = dst = wid - E; }
    const float* pl = xl + (size_t)src * 128;
    const float* pr = xr + (size_t)dst * 128;
    #pragma unroll
    for (int h = 0; h < 4; ++h) {
        int c = h * 32 + lane;
        float v = pl[c] + pr[c];
        v = v > 0.f ? v : SLOPEV * v;          // leaky_relu
        float p = v * att[c];
        #pragma unroll
        for (int off = 16; off > 0; off >>= 1) p += __shfl_xor(p, off, 32);
        if (lane == 0) {
            logit[wid * 4 + h] = p;
            atomicMax(mEnc + dst * 4 + h, encOrd(p));
        }
    }
}

// ---------- decode max, map -inf -> 0 ----------
__global__ void mfix_kernel(const unsigned int* __restrict__ mEnc, float* __restrict__ mf, int n) {
    int i = blockIdx.x * blockDim.x + threadIdx.x;
    if (i < n) {
        float f = decOrd(mEnc[i]);
        mf[i] = isfinite(f) ? f : 0.f;
    }
}

// ---------- E2: exp and denominator ----------
__global__ void e2_kernel(const int* __restrict__ ei,
                          const float* __restrict__ logit, const float* __restrict__ mf,
                          float* __restrict__ ez, float* __restrict__ den,
                          int E, int Nn) {
    int tid = blockIdx.x * blockDim.x + threadIdx.x;
    int T = E + Nn;
    if (tid >= 4 * T) return;
    int e = tid >> 2, h = tid & 3;
    int dst = (e < E) ? ei[E + e] : (e - E);
    float z = __expf(logit[tid] - mf[dst * 4 + h]);
    ez[tid] = z;
    atomicAdd(den + dst * 4 + h, z);
}

// ---------- E3: alpha-weighted scatter-add (wave per edge) ----------
__global__ void e3_kernel(const int* __restrict__ ei,
                          const float* __restrict__ xl,
                          const float* __restrict__ ez, const float* __restrict__ den,
                          float* __restrict__ accDst, int E, int Nn) {
    int wid  = (blockIdx.x * blockDim.x + threadIdx.x) >> 5;
    int lane = threadIdx.x & 31;
    int T = E + Nn;
    if (wid >= T) return;
    int src, dst;
    if (wid < E) { src = ei[wid]; dst = ei[E + wid]; }
    else         { src = dst = wid - E; }
    const float* pl = xl + (size_t)src * 128;
    float* pa = accDst + (size_t)dst * 128;
    #pragma unroll
    for (int h = 0; h < 4; ++h) {
        float a = ez[wid * 4 + h] / (den[dst * 4 + h] + EPSV);
        int c = h * 32 + lane;
        atomicAdd(pa + c, a * pl[c]);
    }
}

// ---------- acc init with summed biases: acc[i][c] = b0[c] + b1[c] + w2*b2[c] ----------
__global__ void acc_init_kernel(float* __restrict__ acc,
                                const float* __restrict__ b0, const float* __restrict__ b1,
                                const float* __restrict__ b2, float w2, int n) {
    int i = blockIdx.x * blockDim.x + threadIdx.x;
    if (i < n) {
        int c = i & 127;
        acc[i] = b0[c] + b1[c] + w2 * b2[c];
    }
}

// ---------- final: mean (divide by fan-in) + relu ----------
__global__ void final_kernel(const float* __restrict__ acc, float* __restrict__ out,
                             int n, int perType) {
    int i = blockIdx.x * blockDim.x + threadIdx.x;
    if (i < n) {
        int t = i / perType;                  // node-type index 0..6
        float cnt = (t < 2) ? 7.f : 3.f;      // q,sub: 7 instances; a_i: 3
        float v = acc[i] / cnt;
        out[i] = v > 0.f ? v : 0.f;
    }
}

extern "C" void kernel_launch(void* const* d_in, const int* in_sizes, int n_in,
                              void* d_out, int out_size, void* d_ws, size_t ws_size,
                              hipStream_t stream) {
    (void)in_sizes; (void)n_in; (void)out_size; (void)ws_size;
    // input layout: [0..6] x_{q,sub,a0..a4}; [7..35] edge indices (2,E) int32
    //               [36 + 4c + {0,1,2,3}] = Wl, Wr, att, bias for conv c (CONV_NAMES order)
    const int PB = 36;
    const int T = NEDGE + N_NODES;

    // ---- carve workspace ----
    size_t off = 0;
    auto carve = [&](size_t bytes) {
        void* p = (char*)d_ws + off;
        off += (bytes + 255) & ~(size_t)255;
        return p;
    };
    float*        xl    = (float*)carve((size_t)N_NODES * 128 * 4);
    float*        xr    = (float*)carve((size_t)N_NODES * 128 * 4);
    float*        logit = (float*)carve((size_t)T * 4 * 4);
    float*        ez    = (float*)carve((size_t)T * 4 * 4);
    unsigned int* mEnc  = (unsigned int*)carve((size_t)N_NODES * 4 * 4);
    float*        mf    = (float*)carve((size_t)N_NODES * 4 * 4);
    float*        den   = (float*)carve((size_t)N_NODES * 4 * 4);
    float*        acc   = (float*)carve((size_t)7 * N_NODES * 128 * 4);
    unsigned short* xb[7];
    for (int i = 0; i < 7; ++i) xb[i] = (unsigned short*)carve((size_t)PAD_ROWS * 128 * 2);
    unsigned short* wt[18];
    for (int i = 0; i < 18; ++i) wt[i] = (unsigned short*)carve((size_t)128 * 128 * 2);

    // ---- pre-passes: bf16 features, bf16 transposed weights ----
    const int nX = N_NODES * 128;
    for (int i = 0; i < 7; ++i)
        cvt_x_kernel<<<(nX + 255) / 256, 256, 0, stream>>>((const float*)d_in[i], xb[i], nX);
    for (int c = 0; c < 9; ++c) {
        cvt_wt_kernel<<<64, 256, 0, stream>>>((const float*)d_in[PB + 4 * c + 0], wt[2 * c + 0]);
        cvt_wt_kernel<<<64, 256, 0, stream>>>((const float*)d_in[PB + 4 * c + 1], wt[2 * c + 1]);
    }

    // ---- edge-type instances: {src_type, dst_type, conv, edge_input_idx} ----
    struct Inst { int s, d, conv, eidx; };
    static const Inst insts[29] = {
        {0,0,0, 7}, {1,1,1, 8},
        {2,2,2, 9}, {3,3,2,10}, {4,4,2,11}, {5,5,2,12}, {6,6,2,13},
        {1,0,3,14}, {0,1,4,15},
        {1,2,5,16}, {1,3,5,17}, {1,4,5,18}, {1,5,5,19}, {1,6,5,20},
        {2,1,6,21}, {3,1,6,22}, {4,1,6,23}, {5,1,6,24}, {6,1,6,25},
        {0,2,7,26}, {0,3,7,27}, {0,4,7,28}, {0,5,7,29}, {0,6,7,30},
        {2,0,8,31}, {3,0,8,32}, {4,0,8,33}, {5,0,8,34}, {6,0,8,35},
    };
    auto bias = [&](int c) { return (const float*)d_in[PB + 4 * c + 3]; };

    // ---- init accumulators with summed biases ----
    const int perType = N_NODES * 128;
    const int gAcc = (perType + 255) / 256;
    // q: q_q + sub_q + 5*a_q
    acc_init_kernel<<<gAcc, 256, 0, stream>>>(acc + 0 * (size_t)perType, bias(0), bias(3), bias(8), 5.f, perType);
    // sub: sub_sub + q_sub + 5*a_sub
    acc_init_kernel<<<gAcc, 256, 0, stream>>>(acc + 1 * (size_t)perType, bias(1), bias(4), bias(6), 5.f, perType);
    // a_i: a_a + sub_a + q_a
    for (int i = 0; i < 5; ++i)
        acc_init_kernel<<<gAcc, 256, 0, stream>>>(acc + (size_t)(2 + i) * perType, bias(2), bias(5), bias(7), 1.f, perType);

    // ---- process each edge-type instance sequentially ----
    const int gGemm = (N_NODES + 127) / 128;      // 79
    const int gWave = (T * 32 + 255) / 256;       // wave-per-edge kernels
    const int gMD   = (4 * N_NODES + 255) / 256;
    const int gEH   = (4 * T + 255) / 256;

    for (int i = 0; i < 29; ++i) {
        const Inst& I = insts[i];
        const float* att = (const float*)d_in[PB + 4 * I.conv + 2];
        const int*   ei  = (const int*)d_in[I.eidx];

        gemm_xw_kernel<<<gGemm, 256, 0, stream>>>(xb[I.s], wt[2 * I.conv + 0], xl, N_NODES);
        gemm_xw_kernel<<<gGemm, 256, 0, stream>>>(xb[I.d], wt[2 * I.conv + 1], xr, N_NODES);
        init_md_kernel<<<gMD, 256, 0, stream>>>(mEnc, den, 4 * N_NODES);
        e1_kernel<<<gWave, 256, 0, stream>>>(ei, xl, xr, att, logit, mEnc, NEDGE, N_NODES);
        mfix_kernel<<<gMD, 256, 0, stream>>>(mEnc, mf, 4 * N_NODES);
        e2_kernel<<<gEH, 256, 0, stream>>>(ei, logit, mf, ez, den, NEDGE, N_NODES);
        e3_kernel<<<gWave, 256, 0, stream>>>(ei, xl, ez, den, acc + (size_t)I.d * perType, NEDGE, N_NODES);
    }

    // ---- finalize: mean + relu into d_out (q, sub, a0..a4 order) ----
    const int nTot = 7 * perType;
    final_kernel<<<(nTot + 255) / 256, 256, 0, stream>>>(acc, (float*)d_out, nTot, perType);
}
